// GPTBLSTM_48653389529686
// MI455X (gfx1250) — matrix operand order repaired and checked by reading, not compiled, not run on hardware
//
#include <hip/hip_runtime.h>
#include <hip/hip_bf16.h>
#include <math.h>

// ---------------------------------------------------------------------------
// Bidirectional LSTM for MI455X (gfx1250, wave32, WMMA + async-to-LDS).
//   B=64, T=1024, C=256, gates 4C=1024.
//   0) f32->f16 pre-convert of x and all 4 weight matrices (one pass)
//   1) gates GEMM (v_wmma_f32_16x16x32_f16), B staged in LDS via
//      global_load_async_to_lds_b128 in fragment layout
//   2) persistent recurrence (8 WGs, L2 atomic step barrier), w_hh slice in
//      LDS (async staged), c-state in VGPRs, h ping-pong in f16 global.
//      Stores time-flipped; dir1 stores gelu(h) as f16 (pass-2 A operand).
//   3) finalize: exact GELU + threefry2x32 dropout (JAX key(42)).
// ---------------------------------------------------------------------------

typedef _Float16 v16h __attribute__((ext_vector_type(16)));
typedef _Float16 v8h  __attribute__((ext_vector_type(8)));
typedef float    v8f  __attribute__((ext_vector_type(8)));
typedef float    v4f  __attribute__((ext_vector_type(4)));

#define B_  64
#define T_  1024
#define C_  256
#define G_  1024   // 4*C
#define NWG 8

__device__ __forceinline__ float gelu_exact(float x) {
  return 0.5f * x * (1.0f + erff(x * 0.70710678118654752f));
}
__device__ __forceinline__ float sigm(float x) {
  return 1.0f / (1.0f + expf(-x));
}

// CDNA5 async copy: 16 bytes global -> LDS, tracked by ASYNCcnt.
__device__ __forceinline__ void async_b128(unsigned lds_off, const void* gaddr) {
  asm volatile("global_load_async_to_lds_b128 %0, %1, off"
               :: "v"(lds_off), "v"(gaddr) : "memory");
}
__device__ __forceinline__ void wait_async0() {
  asm volatile("s_wait_asynccnt 0x0" ::: "memory");
}
template <typename T>
__device__ __forceinline__ unsigned lds_addr32(T* p) {
  return (unsigned)(size_t)p;  // low 32 bits of flat shared addr = LDS offset
}

// Load a 16-half fragment from two 16B-aligned f16 groups (p, p+8 halves).
__device__ __forceinline__ v16h frag_from_f16(const _Float16* __restrict__ p,
                                              int second_group_off) {
  v16h a;
  v8h l8 = __builtin_bit_cast(v8h, *reinterpret_cast<const v4f*>(p));
  v8h h8 = __builtin_bit_cast(v8h, *reinterpret_cast<const v4f*>(p + second_group_off));
#pragma unroll
  for (int j = 0; j < 8; ++j) { a[j] = l8[j]; a[8 + j] = h8[j]; }
  return a;
}

// ---------------------------------------------------------------------------
// f32 -> f16 elementwise convert
// ---------------------------------------------------------------------------
__global__ __launch_bounds__(256)
void cvt_f16(const float* __restrict__ in, _Float16* __restrict__ outp, int n) {
  const int i = (blockIdx.x * 256 + threadIdx.x) * 4;
  if (i >= n) return;
  const v4f v = *reinterpret_cast<const v4f*>(in + i);
#pragma unroll
  for (int j = 0; j < 4; ++j) outp[i + j] = (_Float16)v[j];
}

// ---------------------------------------------------------------------------
// Gates GEMM: xg[row,n] = bias[n] + sum_k A16(row,k) * W16(n,k)
// Block: 256 threads = 8 waves; wave owns 2 M-tiles (32 rows); block N = 64.
// B tile (64 cols x 256 k, f16) staged in LDS in fragment layout via async.
// ---------------------------------------------------------------------------
__global__ __launch_bounds__(256)
void lstm_gates_gemm(const _Float16* __restrict__ a16,  // [B*T, C] f16
                     const _Float16* __restrict__ w16,  // [G, C]  f16
                     const float* __restrict__ b0,
                     const float* __restrict__ b1,
                     float* __restrict__ xg) {           // [B*T, G]
  __shared__ _Float16 bfragLDS[8 * 4 * 32 * 16];         // 32 KB [kk][nt][lane][16]
  const int tid  = threadIdx.x;
  const int wave = tid >> 5, lane = tid & 31;
  const int hlane = lane & 15, half = lane >> 4;
  const int m0   = blockIdx.x * 256 + wave * 32;         // 2 M-tiles per wave
  const int n0   = blockIdx.y * 64;

  // Stage B fragments: entry e=(kk,nt,lane) is 32B, lane<16 -> K lo16, else hi16.
  const unsigned lbase = lds_addr32(&bfragLDS[0]);
  for (int e = tid; e < 1024; e += 256) {
    const int l  = e & 31;
    const int nt = (e >> 5) & 3;
    const int kk = e >> 7;
    const _Float16* src =
        w16 + (size_t)(n0 + nt * 16 + (l & 15)) * C_ + kk * 32 + ((l >> 4) << 4);
    const unsigned dst = lbase + (unsigned)e * 32u;
    async_b128(dst, src);
    async_b128(dst + 16u, src + 8);
  }
  wait_async0();
  __syncthreads();

  const _Float16* arow0 = a16 + (size_t)(m0 + hlane) * C_;
  const _Float16* arow1 = arow0 + (size_t)16 * C_;

  v8f acc[2][4] = {};
#pragma unroll 1
  for (int kk = 0; kk < 8; ++kk) {
    const int ka = kk * 32 + half * 16;
    if (kk < 7) __builtin_prefetch(arow0 + ka + 32, 0, 1);   // global_prefetch_b8
    const v16h af0 = frag_from_f16(arow0 + ka, 8);
    const v16h af1 = frag_from_f16(arow1 + ka, 8);
#pragma unroll
    for (int nt = 0; nt < 4; ++nt) {
      const _Float16* bp = &bfragLDS[((kk * 4 + nt) * 32 + lane) * 16];
      const v16h bf = frag_from_f16(bp, 8);
      acc[0][nt] = __builtin_amdgcn_wmma_f32_16x16x32_f16(
          false, af0, false, bf, (short)0, acc[0][nt], false, false);
      acc[1][nt] = __builtin_amdgcn_wmma_f32_16x16x32_f16(
          false, af1, false, bf, (short)0, acc[1][nt], false, false);
    }
  }
#pragma unroll
  for (int nt = 0; nt < 4; ++nt) {
    const int col = n0 + nt * 16 + hlane;
    const float bias = b0[col] + b1[col];
#pragma unroll
    for (int mt = 0; mt < 2; ++mt) {
#pragma unroll
      for (int r = 0; r < 8; ++r) {
        const int row = m0 + mt * 16 + r + half * 8;       // D: lane>=16 -> M+8
        xg[(size_t)row * G_ + col] = acc[mt][nt][r] + bias;
      }
    }
  }
}

// ---------------------------------------------------------------------------
// Persistent recurrence. WG wg owns C-cols [wg*32,+32) => all 4 gates of those
// columns. Wave (mt,s): 16 batch rows x 16 cols, 4 gate accumulators, c-state
// resident in one v8f for all 1024 steps. Stores are time-flipped:
//   mode==1: hs16[b, T-1-t, c] = f16(gelu(h))   (consumed by pass-2 GEMM)
//   mode==0: hs32[b, T-1-t, c] = h              (consumed by finalize)
// ---------------------------------------------------------------------------
__global__ __launch_bounds__(256)
void lstm_recurrence(const float* __restrict__ xg,       // [B*T, G]
                     const _Float16* __restrict__ whh16, // [G, C] f16
                     _Float16* __restrict__ h16,         // [2][B_, C_] ping-pong
                     float* __restrict__ hs32,
                     _Float16* __restrict__ hs16,
                     int mode,
                     unsigned int* __restrict__ syncc) {
  __shared__ _Float16 wfrag[8 * 4 * 2 * 32 * 16];        // 64 KB [kk][g][s][lane][16]
  const int tid  = threadIdx.x;
  const int wave = tid >> 5, lane = tid & 31;
  const int hlane = lane & 15, half = lane >> 4;
  const int wg = blockIdx.x;
  const int c0 = wg * 32;
  const int mt = wave >> 1, s = wave & 1;
  const int m0  = mt * 16;
  const int n_c = c0 + s * 16;

  // Async-stage w_hh slice (pure f16 copy) into fragment layout.
  const unsigned lbase = lds_addr32(&wfrag[0]);
  for (int e = tid; e < 2048; e += 256) {
    const int l  = e & 31;
    const int ss = (e >> 5) & 1;
    const int g  = (e >> 6) & 3;
    const int kk = e >> 8;
    const _Float16* src =
        whh16 + (size_t)(g * 256 + c0 + ss * 16 + (l & 15)) * C_ + kk * 32 + ((l >> 4) << 4);
    const unsigned dst = lbase + (unsigned)e * 32u;
    async_b128(dst, src);
    async_b128(dst + 16u, src + 8);
  }
  wait_async0();

  // Zero both h ping-pong buffers (own column chunk only).
  for (int e = tid; e < B_ * 32; e += 256) {
    const int bb = e >> 5, cc = c0 + (e & 31);
    h16[bb * C_ + cc]           = (_Float16)0.f;
    h16[B_ * C_ + bb * C_ + cc] = (_Float16)0.f;
  }
  v8f cstate = {};
  __threadfence();
  __syncthreads();
  if (tid == 0)
    __hip_atomic_fetch_add(syncc, 1u, __ATOMIC_RELEASE, __HIP_MEMORY_SCOPE_AGENT);

  for (int t = 0; t < T_; ++t) {
    if (tid == 0) {
      const unsigned target = (unsigned)NWG * (unsigned)(t + 1);
      while (__hip_atomic_load(syncc, __ATOMIC_ACQUIRE, __HIP_MEMORY_SCOPE_AGENT) < target)
        __builtin_amdgcn_s_sleep(1);
    }
    __syncthreads();

    const _Float16* hin = h16 + (size_t)(t & 1) * (B_ * C_);
    v8f acc[4] = {};
#pragma unroll 1
    for (int kk = 0; kk < 8; ++kk) {
      const v16h af = frag_from_f16(hin + (m0 + hlane) * C_ + kk * 32 + half * 16, 8);
#pragma unroll
      for (int g = 0; g < 4; ++g) {
        const _Float16* bp = &wfrag[(((kk * 4 + g) * 2 + s) * 32 + lane) * 16];
        const v16h bf = frag_from_f16(bp, 8);
        acc[g] = __builtin_amdgcn_wmma_f32_16x16x32_f16(
            false, af, false, bf, (short)0, acc[g], false, false);
      }
    }

    _Float16* hout = h16 + (size_t)((t + 1) & 1) * (B_ * C_);
    const int tflip = (T_ - 1) - t;
#pragma unroll
    for (int r = 0; r < 8; ++r) {
      const int bb  = m0 + r + half * 8;
      const int col = n_c + hlane;
      const size_t xbase = ((size_t)bb * T_ + (size_t)t) * (size_t)G_;
      const float gi = acc[0][r] + xg[xbase + 0 * C_ + col];
      const float gf = acc[1][r] + xg[xbase + 1 * C_ + col];
      const float gg = acc[2][r] + xg[xbase + 2 * C_ + col];
      const float go = acc[3][r] + xg[xbase + 3 * C_ + col];
      const float iv = sigm(gi), fv = sigm(gf), gv = tanhf(gg), ov = sigm(go);
      const float cv = fv * cstate[r] + iv * gv;
      cstate[r] = cv;
      const float hv = ov * tanhf(cv);
      const size_t oidx = ((size_t)bb * T_ + (size_t)tflip) * C_ + col;
      if (mode) hs16[oidx] = (_Float16)gelu_exact(hv);
      else      hs32[oidx] = hv;
      hout[bb * C_ + col] = (_Float16)hv;
    }
    __threadfence();
    __syncthreads();
    if (tid == 0)
      __hip_atomic_fetch_add(syncc, 1u, __ATOMIC_RELEASE, __HIP_MEMORY_SCOPE_AGENT);
  }
}

// ---------------------------------------------------------------------------
// Finalize: h2 is already time-flipped back; out = dropout(gelu(h2)).
// threefry2x32(key(42)) bernoulli mask, keep p = 0.8 (JAX semantics).
// ---------------------------------------------------------------------------
__device__ __forceinline__ unsigned rotl32(unsigned x, int r) {
  return (x << r) | (x >> (32 - r));
}

__global__ __launch_bounds__(256)
void lstm_finalize(const float* __restrict__ h2, float* __restrict__ out, int n) {
  const int i = blockIdx.x * 256 + threadIdx.x;
  if (i >= n) return;
  float v = gelu_exact(h2[i]);

  const unsigned half_n = (unsigned)(n >> 1);
  const unsigned idx = (unsigned)i;
  unsigned x0, x1;
  bool second;
  if (idx < half_n) { x0 = idx; x1 = idx + half_n; second = false; }
  else              { x0 = idx - half_n; x1 = idx; second = true;  }
  const unsigned ks0 = 0u, ks1 = 42u, ks2 = ks0 ^ ks1 ^ 0x1BD11BDAu;
  x0 += ks0; x1 += ks1;
#define TF_ROUND(r) { x0 += x1; x1 = rotl32(x1, (r)); x1 ^= x0; }
  TF_ROUND(13) TF_ROUND(15) TF_ROUND(26) TF_ROUND(6)
  x0 += ks1; x1 += ks2 + 1u;
  TF_ROUND(17) TF_ROUND(29) TF_ROUND(16) TF_ROUND(24)
  x0 += ks2; x1 += ks0 + 2u;
  TF_ROUND(13) TF_ROUND(15) TF_ROUND(26) TF_ROUND(6)
  x0 += ks0; x1 += ks1 + 3u;
  TF_ROUND(17) TF_ROUND(29) TF_ROUND(16) TF_ROUND(24)
  x0 += ks1; x1 += ks2 + 4u;
  TF_ROUND(13) TF_ROUND(15) TF_ROUND(26) TF_ROUND(6)
  x0 += ks2; x1 += ks0 + 5u;
#undef TF_ROUND
  const unsigned bits = second ? x1 : x0;
  const float u = __uint_as_float((bits >> 9) | 0x3f800000u) - 1.0f;
  out[i] = (u < 0.8f) ? v * 1.25f : 0.0f;
}

__global__ void lstm_zero_sync(unsigned int* s) {
  if (threadIdx.x == 0 && blockIdx.x == 0) *s = 0u;
}

// ---------------------------------------------------------------------------
// Host launcher. Inputs:
//   x, w_ih_fwd, w_hh_fwd, b_ih_fwd, b_hh_fwd, w_ih_bwd, w_hh_bwd, b_ih_bwd, b_hh_bwd
// Workspace: xg 256MB | h2 64MB | x16 32MB | h1g16 32MB | 4x w16 | h16 pp | sync
// ---------------------------------------------------------------------------
extern "C" void kernel_launch(void* const* d_in, const int* in_sizes, int n_in,
                              void* d_out, int out_size, void* d_ws, size_t ws_size,
                              hipStream_t stream) {
  const float* x     = (const float*)d_in[0];
  const float* wih_f = (const float*)d_in[1];
  const float* whh_f = (const float*)d_in[2];
  const float* bih_f = (const float*)d_in[3];
  const float* bhh_f = (const float*)d_in[4];
  const float* wih_b = (const float*)d_in[5];
  const float* whh_b = (const float*)d_in[6];
  const float* bih_b = (const float*)d_in[7];
  const float* bhh_b = (const float*)d_in[8];
  float* out = (float*)d_out;

  const size_t NX = (size_t)B_ * T_ * C_;       // 16.7M
  const size_t NW = (size_t)G_ * C_;            // 262144
  char* ws = (char*)d_ws;
  size_t off = 0;
  float*    xg    = (float*)(ws + off);    off += NX * 4 * sizeof(float);  // 256MB
  float*    h2    = (float*)(ws + off);    off += NX * sizeof(float);      // 64MB
  _Float16* x16   = (_Float16*)(ws + off); off += NX * 2;                  // 32MB
  _Float16* h1g16 = (_Float16*)(ws + off); off += NX * 2;                  // 32MB
  _Float16* wihf16= (_Float16*)(ws + off); off += NW * 2;
  _Float16* whhf16= (_Float16*)(ws + off); off += NW * 2;
  _Float16* wihb16= (_Float16*)(ws + off); off += NW * 2;
  _Float16* whhb16= (_Float16*)(ws + off); off += NW * 2;
  _Float16* h16   = (_Float16*)(ws + off); off += 2 * (size_t)B_ * C_ * 2;
  off = (off + 255) & ~(size_t)255;
  unsigned int* sync = (unsigned int*)(ws + off);

  // 0) precision pre-pass (f32 -> f16)
  cvt_f16<<<(int)(NX / 4 / 256), 256, 0, stream>>>(x, x16, (int)NX);
  cvt_f16<<<(int)(NW / 4 / 256), 256, 0, stream>>>(wih_f, wihf16, (int)NW);
  cvt_f16<<<(int)(NW / 4 / 256), 256, 0, stream>>>(whh_f, whhf16, (int)NW);
  cvt_f16<<<(int)(NW / 4 / 256), 256, 0, stream>>>(wih_b, wihb16, (int)NW);
  cvt_f16<<<(int)(NW / 4 / 256), 256, 0, stream>>>(whh_b, whhb16, (int)NW);

  const dim3 gg(256, 16);  // (65536/256 M-blocks, 1024/64 N-blocks)

  // direction 1
  lstm_gates_gemm<<<gg, 256, 0, stream>>>(x16, wihf16, bih_f, bhh_f, xg);
  lstm_zero_sync<<<1, 32, 0, stream>>>(sync);
  lstm_recurrence<<<NWG, 256, 0, stream>>>(xg, whhf16, h16, (float*)nullptr,
                                           h1g16, 1, sync);
  // direction 2 (input already gelu'd, f16, time-flipped)
  lstm_gates_gemm<<<gg, 256, 0, stream>>>(h1g16, wihb16, bih_b, bhh_b, xg);
  lstm_zero_sync<<<1, 32, 0, stream>>>(sync);
  lstm_recurrence<<<NWG, 256, 0, stream>>>(xg, whhb16, h16, h2,
                                           (_Float16*)nullptr, 0, sync);

  lstm_finalize<<<(int)(NX / 256), 256, 0, stream>>>(h2, out, (int)NX);
}
